// Qwen2MoeSparseMoeBlock_12378095747250
// MI455X (gfx1250) — matrix operand order repaired and checked by reading, not compiled, not run on hardware
//
#include <hip/hip_runtime.h>
#include <stdint.h>

#define H_DIM  2048
#define N_EXP  16
#define I_EXPT 1408
#define I_SHRD 5632
#define M_TOK  4096
#define TOPK   4

typedef __attribute__((ext_vector_type(16))) __bf16 v16bf;
typedef __attribute__((ext_vector_type(8)))  float  v8f;

struct __align__(16) U4 { unsigned x, y, z, w; };
union FragU { v16bf v; U4 q[2]; };

// exact RNE (used for sparse epilogue stores)
__device__ __forceinline__ unsigned short f2bf(float f) {
  unsigned u = __float_as_uint(f);
  u += 0x7FFFu + ((u >> 16) & 1u);
  return (unsigned short)(u >> 16);
}
// fast pack: round-half-up + byte-perm -> 3 VALU per pair
__device__ __forceinline__ unsigned rb(float f) { return __float_as_uint(f) + 0x8000u; }
__device__ __forceinline__ unsigned pack2(float a, float b) {
  // result low16 = bf16(a), high16 = bf16(b)  (v_perm_b32 selects high bytes)
  return __builtin_amdgcn_perm(rb(b), rb(a), 0x07060302u);
}
__device__ __forceinline__ float sigm(float x) { return 1.f / (1.f + __expf(-x)); }

// CDNA5 async global->LDS copy (ASYNCcnt-tracked), 16B per lane
__device__ __forceinline__ void async_load_b128(unsigned lds_off, const void* gptr) {
  asm volatile("global_load_async_to_lds_b128 %0, %1, off"
               :: "v"(lds_off), "v"((unsigned long long)(size_t)gptr)
               : "memory");
}
__device__ __forceinline__ void wait_async0() {
  asm volatile("s_wait_asynccnt 0x0" ::: "memory");
}
__device__ __forceinline__ unsigned lds_off(const void* p) {
  return (unsigned)(size_t)p;  // LDS aperture: low 32 bits of flat addr = LDS offset
}

// A-matrix 16x32 bf16 frag per ISA: lane half picks K chunks [8h,8h+8) and [16+8h,24+8h)
__device__ __forceinline__ v16bf frag_a(const unsigned short* row, int half) {
  FragU f; const U4* p = (const U4*)row;
  f.q[0] = p[half]; f.q[1] = p[half + 2];
  return f.v;
}
// B-matrix 32x16 bf16 frag: lane half picks K chunks [8h,8h+8) and [16+8h,24+8h)
__device__ __forceinline__ v16bf frag_b(const unsigned short* row, int half) {
  FragU f; const U4* p = (const U4*)row;
  f.q[0] = p[half]; f.q[1] = p[half + 2];
  return f.v;
}

__device__ __forceinline__ void pack_store16(unsigned short* dst, float4 a, float4 b,
                                             float4 c, float4 d) {
  U4 q0, q1;
  q0.x = pack2(a.x, a.y); q0.y = pack2(a.z, a.w);
  q0.z = pack2(b.x, b.y); q0.w = pack2(b.z, b.w);
  q1.x = pack2(c.x, c.y); q1.y = pack2(c.z, c.w);
  q1.z = pack2(d.x, d.y); q1.w = pack2(d.z, d.w);
  *(U4*)dst = q0;
  *(U4*)(dst + 8) = q1;
}

// ---------------- fp32 -> bf16 conversion of activations ----------------
__global__ void cvt_x_kernel(const float* __restrict__ x, unsigned short* __restrict__ xb, int n8) {
  int t = blockIdx.x * blockDim.x + threadIdx.x;
  if (t >= n8) return;
  const float4* p = (const float4*)x + (size_t)t * 2;
  float4 a = p[0], b = p[1];
  U4 o;
  o.x = pack2(a.x, a.y); o.y = pack2(a.z, a.w);
  o.z = pack2(b.x, b.y); o.w = pack2(b.z, b.w);
  ((U4*)xb)[t] = o;
}

// ---------------- router: logits + softmax + top-4 + shared sigmoid gate ----------------
__global__ void router_kernel(const float* __restrict__ x, const float* __restrict__ gate_w,
                              const float* __restrict__ sgate_w, float* __restrict__ sg,
                              int* __restrict__ topk_id, float* __restrict__ topk_w,
                              int* __restrict__ counts) {
  int m = blockIdx.x, lane = threadIdx.x;
  float acc[N_EXP + 1];
#pragma unroll
  for (int j = 0; j <= N_EXP; j++) acc[j] = 0.f;
  const float* xr = x + (size_t)m * H_DIM;
  for (int k = lane; k < H_DIM; k += 32) {
    float xv = xr[k];
#pragma unroll
    for (int e = 0; e < N_EXP; e++) acc[e] += xv * gate_w[e * H_DIM + k];
    acc[N_EXP] += xv * sgate_w[k];
  }
#pragma unroll
  for (int j = 0; j <= N_EXP; j++)
    for (int o = 16; o > 0; o >>= 1) acc[j] += __shfl_xor(acc[j], o, 32);
  if (lane == 0) {
    float mx = acc[0];
#pragma unroll
    for (int e = 1; e < N_EXP; e++) mx = fmaxf(mx, acc[e]);
    float ex[N_EXP], s = 0.f;
#pragma unroll
    for (int e = 0; e < N_EXP; e++) { ex[e] = __expf(acc[e] - mx); s += ex[e]; }
    float inv = 1.f / s;
#pragma unroll
    for (int j = 0; j < TOPK; j++) {
      int bi = 0; float bv = -1.f;
#pragma unroll
      for (int e = 0; e < N_EXP; e++) if (ex[e] > bv) { bv = ex[e]; bi = e; }
      topk_id[m * TOPK + j] = bi;
      topk_w[m * TOPK + j] = bv * inv;
      ex[bi] = -1.f;
      atomicAdd(&counts[bi], 1);
    }
    sg[m] = sigm(acc[N_EXP]);
  }
}

__global__ void prefix_kernel(const int* __restrict__ counts, int* __restrict__ offs) {
  if (threadIdx.x == 0) {
    int s = 0;
    for (int e = 0; e < N_EXP; e++) { offs[e] = s; s += counts[e]; }
    offs[N_EXP] = s;
  }
}

// deterministic ballot-based compaction: one wave per expert
__global__ void compact_kernel(const int* __restrict__ topk_id, const float* __restrict__ topk_w,
                               const int* __restrict__ offs, int* __restrict__ tokc,
                               float* __restrict__ coefc) {
  int e = blockIdx.x, lane = threadIdx.x;
  int base = offs[e], pos = 0;
  for (int c = 0; c < M_TOK; c += 32) {
    int t = c + lane;
    bool hit = false; float w = 0.f;
#pragma unroll
    for (int j = 0; j < TOPK; j++)
      if (topk_id[t * TOPK + j] == e) { hit = true; w = topk_w[t * TOPK + j]; }
    unsigned mask = (unsigned)__ballot(hit);
    int p = __popc(mask & ((1u << lane) - 1u));
    if (hit) { tokc[base + pos + p] = t; coefc[base + pos + p] = w; }
    pos += __popc(mask);
  }
}

// ---------------- gated up-projection GEMM: h = silu(X Wg^T) * (X Wu^T) ----------------
// block tile 64x128; 8 waves as 4(M) x 2(N); double-buffered LDS, 1 barrier/K-step;
// A tile via async global->LDS, B tiles prefetched to regs one step ahead;
// all 8 B fragments kept live so DS loads batch ahead of the WMMA burst.
__global__ __launch_bounds__(256) void up_gemm_kernel(
    const unsigned short* __restrict__ xb,   // [M,H] bf16
    const float* __restrict__ Wall,          // expert: w13 base [E,2I,H]; shared: [2I,H]
    const int* __restrict__ offs, const int* __restrict__ tokc,
    unsigned short* __restrict__ hout,       // expert: he rows=routed; shared: hs rows=M
    int Icols, int useGather) {
  int e = blockIdx.z;
  int rowBase = 0, cnt = M_TOK;
  const float* W = Wall;
  if (useGather) {
    int o0 = offs[e];
    cnt = offs[e + 1] - o0;
    rowBase = o0;
    if ((int)blockIdx.y * 64 >= cnt) return;
    W = Wall + (size_t)e * 2 * (size_t)Icols * H_DIM;
  }
  int m0 = blockIdx.y * 64, n0 = blockIdx.x * 128;

  __shared__ __align__(16) unsigned short Ald[2][64 * 32];
  __shared__ __align__(16) unsigned short Bgl[2][128 * 32];
  __shared__ __align__(16) unsigned short Bul[2][128 * 32];

  int tid = threadIdx.x, lane = tid & 31, wave = tid >> 5;
  int mwave = wave & 3, nwave = wave >> 2;
  int half = lane >> 4, l16 = lane & 15;

  // A staging role: 64 rows x 4 segs of 8 halfs (16B async per thread)
  int arow = tid >> 2, aseg = tid & 3;
  int r = m0 + arow;
  int tokenA = 0;  // padding rows load token 0; their outputs are discarded
  if (r < cnt) tokenA = useGather ? tokc[rowBase + r] : r;
  const unsigned short* agp = xb + (size_t)tokenA * H_DIM + aseg * 8;
  unsigned ldsA0 = lds_off(&Ald[0][arow * 32 + aseg * 8]);
  unsigned ldsA1 = lds_off(&Ald[1][arow * 32 + aseg * 8]);

  // B staging role: 128 rows x 2 segs of 16 floats
  int brow = tid >> 1, bseg = tid & 1;
  const float* wg = W + (size_t)(n0 + brow) * H_DIM + bseg * 16;
  const float* wu = W + (size_t)(n0 + brow + Icols) * H_DIM + bseg * 16;
  unsigned short* bgp[2] = { &Bgl[0][brow * 32 + bseg * 16], &Bgl[1][brow * 32 + bseg * 16] };
  unsigned short* bup[2] = { &Bul[0][brow * 32 + bseg * 16], &Bul[1][brow * 32 + bseg * 16] };

  v8f accg[4] = {}, accu[4] = {};

  // prologue: async A(k=0) in flight, B(k=0) into regs
  async_load_b128(ldsA0, agp);
  float4 g0 = ((const float4*)wg)[0], g1 = ((const float4*)wg)[1],
         g2 = ((const float4*)wg)[2], g3 = ((const float4*)wg)[3];
  float4 u0 = ((const float4*)wu)[0], u1 = ((const float4*)wu)[1],
         u2 = ((const float4*)wu)[2], u3 = ((const float4*)wu)[3];

  for (int k0 = 0; k0 < H_DIM; k0 += 32) {
    int p = (k0 >> 5) & 1;
    pack_store16(bgp[p], g0, g1, g2, g3);
    pack_store16(bup[p], u0, u1, u2, u3);
    wait_async0();       // our A(k0) landed in LDS
    __syncthreads();     // tiles(k0) visible; buffers 1-p free for refill

    if (k0 + 32 < H_DIM) {
      async_load_b128(p ? ldsA0 : ldsA1, agp + (k0 + 32));
      const float4* pg = (const float4*)(wg + k0 + 32);
      g0 = pg[0]; g1 = pg[1]; g2 = pg[2]; g3 = pg[3];
      const float4* pu = (const float4*)(wu + k0 + 32);
      u0 = pu[0]; u1 = pu[1]; u2 = pu[2]; u3 = pu[3];
    }

    // preload ALL fragments (kept simultaneously live -> batched ds_loads),
    // then issue the WMMA burst
    const unsigned short* bgbase = &Bgl[p][(nwave * 64 + l16) * 32];
    const unsigned short* bubase = &Bul[p][(nwave * 64 + l16) * 32];
    v16bf af  = frag_a(&Ald[p][(mwave * 16 + l16) * 32], half);
    v16bf bg0 = frag_b(bgbase + 0 * 16 * 32, half);
    v16bf bu0 = frag_b(bubase + 0 * 16 * 32, half);
    v16bf bg1 = frag_b(bgbase + 1 * 16 * 32, half);
    v16bf bu1 = frag_b(bubase + 1 * 16 * 32, half);
    v16bf bg2 = frag_b(bgbase + 2 * 16 * 32, half);
    v16bf bu2 = frag_b(bubase + 2 * 16 * 32, half);
    v16bf bg3 = frag_b(bgbase + 3 * 16 * 32, half);
    v16bf bu3 = frag_b(bubase + 3 * 16 * 32, half);

    accg[0] = __builtin_amdgcn_wmma_f32_16x16x32_bf16(false, af, false, bg0,
                                                      (short)0, accg[0], false, false);
    accu[0] = __builtin_amdgcn_wmma_f32_16x16x32_bf16(false, af, false, bu0,
                                                      (short)0, accu[0], false, false);
    accg[1] = __builtin_amdgcn_wmma_f32_16x16x32_bf16(false, af, false, bg1,
                                                      (short)0, accg[1], false, false);
    accu[1] = __builtin_amdgcn_wmma_f32_16x16x32_bf16(false, af, false, bu1,
                                                      (short)0, accu[1], false, false);
    accg[2] = __builtin_amdgcn_wmma_f32_16x16x32_bf16(false, af, false, bg2,
                                                      (short)0, accg[2], false, false);
    accu[2] = __builtin_amdgcn_wmma_f32_16x16x32_bf16(false, af, false, bu2,
                                                      (short)0, accu[2], false, false);
    accg[3] = __builtin_amdgcn_wmma_f32_16x16x32_bf16(false, af, false, bg3,
                                                      (short)0, accg[3], false, false);
    accu[3] = __builtin_amdgcn_wmma_f32_16x16x32_bf16(false, af, false, bu3,
                                                      (short)0, accu[3], false, false);
  }

#pragma unroll
  for (int v = 0; v < 8; v++) {
    int rloc = mwave * 16 + 8 * half + v;
    int rr = m0 + rloc;
    if (rr >= cnt) continue;
    size_t outRow = (size_t)(rowBase + rr) * Icols;
#pragma unroll
    for (int i = 0; i < 4; i++) {
      int c = n0 + nwave * 64 + i * 16 + l16;
      float g = accg[i][v], u = accu[i][v];
      hout[outRow + c] = f2bf(g * sigm(g) * u);
    }
  }
}

// ---------------- down-projection GEMM: out (+)= (h W2^T) * scale ----------------
__global__ __launch_bounds__(256) void down_gemm_kernel(
    const unsigned short* __restrict__ hin,  // [rows, Kdim] bf16
    const float* __restrict__ Wall,          // expert: w2 base [E,H,I]; shared: [H,I_SH]
    const int* __restrict__ offs, const int* __restrict__ tokc,
    const float* __restrict__ coefc, const float* __restrict__ sg,
    float* __restrict__ out, int Kdim, int useGather) {
  int e = blockIdx.z;
  int rowBase = 0, cnt = M_TOK;
  const float* W = Wall;
  if (useGather) {
    int o0 = offs[e];
    cnt = offs[e + 1] - o0;
    rowBase = o0;
    if ((int)blockIdx.y * 64 >= cnt) return;
    W = Wall + (size_t)e * (size_t)H_DIM * Kdim;
  }
  int m0 = blockIdx.y * 64, n0 = blockIdx.x * 128;

  __shared__ __align__(16) unsigned short Ald[2][64 * 32];
  __shared__ __align__(16) unsigned short Bld[2][128 * 32];

  int tid = threadIdx.x, lane = tid & 31, wave = tid >> 5;
  int mwave = wave & 3, nwave = wave >> 2;
  int half = lane >> 4, l16 = lane & 15;

  int arow = tid >> 2, aseg = tid & 3;
  int r = m0 + arow;
  int ra = (r < cnt) ? r : 0;  // padding rows read row 0; outputs discarded
  const unsigned short* agp = hin + (size_t)(rowBase + ra) * Kdim + aseg * 8;
  unsigned ldsA0 = lds_off(&Ald[0][arow * 32 + aseg * 8]);
  unsigned ldsA1 = lds_off(&Ald[1][arow * 32 + aseg * 8]);

  int brow = tid >> 1, bseg = tid & 1;
  const float* wr = W + (size_t)(n0 + brow) * Kdim + bseg * 16;
  unsigned short* blp[2] = { &Bld[0][brow * 32 + bseg * 16], &Bld[1][brow * 32 + bseg * 16] };

  v8f acc[4] = {};

  async_load_b128(ldsA0, agp);
  float4 b0 = ((const float4*)wr)[0], b1 = ((const float4*)wr)[1],
         b2 = ((const float4*)wr)[2], b3 = ((const float4*)wr)[3];

  for (int k0 = 0; k0 < Kdim; k0 += 32) {
    int p = (k0 >> 5) & 1;
    pack_store16(blp[p], b0, b1, b2, b3);
    wait_async0();
    __syncthreads();

    if (k0 + 32 < Kdim) {
      async_load_b128(p ? ldsA0 : ldsA1, agp + (k0 + 32));
      const float4* pb = (const float4*)(wr + k0 + 32);
      b0 = pb[0]; b1 = pb[1]; b2 = pb[2]; b3 = pb[3];
    }

    const unsigned short* bbase = &Bld[p][(nwave * 64 + l16) * 32];
    v16bf af  = frag_a(&Ald[p][(mwave * 16 + l16) * 32], half);
    v16bf fb0 = frag_b(bbase + 0 * 16 * 32, half);
    v16bf fb1 = frag_b(bbase + 1 * 16 * 32, half);
    v16bf fb2 = frag_b(bbase + 2 * 16 * 32, half);
    v16bf fb3 = frag_b(bbase + 3 * 16 * 32, half);

    acc[0] = __builtin_amdgcn_wmma_f32_16x16x32_bf16(false, af, false, fb0,
                                                     (short)0, acc[0], false, false);
    acc[1] = __builtin_amdgcn_wmma_f32_16x16x32_bf16(false, af, false, fb1,
                                                     (short)0, acc[1], false, false);
    acc[2] = __builtin_amdgcn_wmma_f32_16x16x32_bf16(false, af, false, fb2,
                                                     (short)0, acc[2], false, false);
    acc[3] = __builtin_amdgcn_wmma_f32_16x16x32_bf16(false, af, false, fb3,
                                                     (short)0, acc[3], false, false);
  }

#pragma unroll
  for (int v = 0; v < 8; v++) {
    int rloc = mwave * 16 + 8 * half + v;
    int rr = m0 + rloc;
    if (rr >= cnt) continue;
    if (useGather) {
      int token = tokc[rowBase + rr];
      float scale = coefc[rowBase + rr];
#pragma unroll
      for (int i = 0; i < 4; i++) {
        int c = n0 + nwave * 64 + i * 16 + l16;
        atomicAdd(&out[(size_t)token * H_DIM + c], acc[i][v] * scale);
      }
    } else {
      float scale = sg[rr];
#pragma unroll
      for (int i = 0; i < 4; i++) {
        int c = n0 + nwave * 64 + i * 16 + l16;
        out[(size_t)rr * H_DIM + c] = acc[i][v] * scale;
      }
    }
  }
}

extern "C" void kernel_launch(void* const* d_in, const int* in_sizes, int n_in,
                              void* d_out, int out_size, void* d_ws, size_t ws_size,
                              hipStream_t stream) {
  const float* x       = (const float*)d_in[0];
  const float* gate_w  = (const float*)d_in[1];
  const float* sgate_w = (const float*)d_in[2];
  const float* w13     = (const float*)d_in[3];
  const float* w2      = (const float*)d_in[4];
  const float* sh_gu   = (const float*)d_in[5];
  const float* sh_d    = (const float*)d_in[6];
  float* out = (float*)d_out;

  char* ws = (char*)d_ws;
  size_t off = 0;
  auto alloc = [&](size_t bytes) { size_t p = off; off += (bytes + 255) & ~(size_t)255; return p; };
  unsigned short* xb    = (unsigned short*)(ws + alloc((size_t)M_TOK * H_DIM * 2));
  unsigned short* hs    = (unsigned short*)(ws + alloc((size_t)M_TOK * I_SHRD * 2));
  unsigned short* he    = (unsigned short*)(ws + alloc((size_t)M_TOK * TOPK * I_EXPT * 2));
  float* sg             = (float*)(ws + alloc((size_t)M_TOK * 4));
  int* topk_id          = (int*)(ws + alloc((size_t)M_TOK * TOPK * 4));
  float* topk_w         = (float*)(ws + alloc((size_t)M_TOK * TOPK * 4));
  int* counts           = (int*)(ws + alloc(64));
  int* offs             = (int*)(ws + alloc(128));
  int* tokc             = (int*)(ws + alloc((size_t)M_TOK * TOPK * 4));
  float* coefc          = (float*)(ws + alloc((size_t)M_TOK * TOPK * 4));

  hipMemsetAsync(counts, 0, 64, stream);

  int n8 = M_TOK * H_DIM / 8;
  cvt_x_kernel<<<(n8 + 255) / 256, 256, 0, stream>>>(x, xb, n8);
  router_kernel<<<M_TOK, 32, 0, stream>>>(x, gate_w, sgate_w, sg, topk_id, topk_w, counts);
  prefix_kernel<<<1, 32, 0, stream>>>(counts, offs);
  compact_kernel<<<N_EXP, 32, 0, stream>>>(topk_id, topk_w, offs, tokc, coefc);

  // shared expert up then routed experts up
  up_gemm_kernel<<<dim3(I_SHRD / 128, M_TOK / 64, 1), 256, 0, stream>>>(
      xb, sh_gu, nullptr, nullptr, hs, I_SHRD, 0);
  up_gemm_kernel<<<dim3(I_EXPT / 128, M_TOK / 64, N_EXP), 256, 0, stream>>>(
      xb, w13, offs, tokc, he, I_EXPT, 1);

  // shared down writes out (covers every element), then expert down atomically adds
  down_gemm_kernel<<<dim3(H_DIM / 128, M_TOK / 64, 1), 256, 0, stream>>>(
      hs, sh_d, nullptr, nullptr, nullptr, sg, out, I_SHRD, 0);
  down_gemm_kernel<<<dim3(H_DIM / 128, M_TOK / 64, N_EXP), 256, 0, stream>>>(
      he, w2, offs, tokc, coefc, nullptr, out, I_EXPT, 1);
}
